// SoftAgg_20134806683721
// MI455X (gfx1250) — matrix operand order repaired and checked by reading, not compiled, not run on hardware
//
#include <hip/hip_runtime.h>

// SoftAgg on MI455X (gfx1250, wave32, WMMA f32_16x16x32_f16).
// Pipeline: prep (zero accum + fp32->f16 weights) -> main (2 fused WMMA GEMMs
// + exp + segment atomics) -> seg GEMM on [1024,384] -> gather.
// Round 2: cap K-loop unroll to stop VGPR blow-up / scratch spills seen in
// round-1 asm; add NT cache hints on the stream-once x read and out write.

typedef __attribute__((ext_vector_type(16))) _Float16 v16h;
typedef __attribute__((ext_vector_type(8)))  _Float16 v8h;
typedef __attribute__((ext_vector_type(4)))  _Float16 v4h;
typedef __attribute__((ext_vector_type(8)))  float    v8f;
typedef __attribute__((ext_vector_type(4)))  float    v4f;

#define DIMC   384
#define NSEG   1024
#define KSTEPS (DIMC / 32)   // 12
#define NTILES (DIMC / 16)   // 24
#define MROWS  64            // rows per workgroup in main kernel
#define SA     (DIMC + 8)    // padded LDS row stride (halves) to dodge bank conflicts

// ---------------------------------------------------------------- prep ------
__global__ void __launch_bounds__(256)
prep_kernel(const float* __restrict__ fw, const float* __restrict__ gw,
            const float* __restrict__ hw,
            float* __restrict__ Dsum, float* __restrict__ Psum,
            _Float16* __restrict__ fw16, _Float16* __restrict__ gw16,
            _Float16* __restrict__ hw16) {
  const int i      = blockIdx.x * blockDim.x + threadIdx.x;
  const int stride = gridDim.x * blockDim.x;
  for (int j = i; j < NSEG * DIMC; j += stride) { Dsum[j] = 0.f; Psum[j] = 0.f; }
  for (int j = i; j < DIMC * DIMC; j += stride) {
    fw16[j] = (_Float16)fw[j];
    gw16[j] = (_Float16)gw[j];
    hw16[j] = (_Float16)hw[j];
  }
}

// ---------------------------------------------------------------- main ------
// One workgroup = 64 rows of x. Stage x->f16 into LDS once, then 8 waves
// each own 3 N-tiles; B fragments are reused across 4 M-subtiles in regs.
__global__ void __launch_bounds__(256)
main_kernel(const float* __restrict__ x, const int* __restrict__ ix,
            const float* __restrict__ fb, const float* __restrict__ gb,
            const _Float16* __restrict__ fw16, const _Float16* __restrict__ gw16,
            float* __restrict__ Dsum, float* __restrict__ Psum) {
  __shared__ _Float16 As[MROWS * SA];
  __shared__ int segs[MROWS];

  const int tid  = threadIdx.x;
  const int lane = tid & 31;
  const int wave = tid >> 5;
  const int row0 = blockIdx.x * MROWS;

  // ---- stage x (fp32) -> f16 LDS; x is read exactly once -> NT loads so the
  // 96 MB stream does not evict weights/Dsum/Psum from L2.
  const float* xsrc = x + (size_t)row0 * DIMC;
  for (int i = tid; i < MROWS * DIMC / 4; i += 256) {
    v4f v = __builtin_nontemporal_load((const v4f*)xsrc + i);
    const int e = i * 4;
    const int r = e / DIMC;
    const int c = e % DIMC;                 // DIMC % 4 == 0, never crosses rows
    v4h h;
    h[0] = (_Float16)v[0]; h[1] = (_Float16)v[1];
    h[2] = (_Float16)v[2]; h[3] = (_Float16)v[3];
    *(v4h*)&As[r * SA + c] = h;
  }
  if (tid < MROWS) segs[tid] = ix[row0 + tid];
  __syncthreads();

  const int half = lane >> 4;
  const int l16  = lane & 15;

#pragma clang loop unroll(disable)
  for (int t = 0; t < 3; ++t) {
    const int nt  = wave * 3 + t;          // 8 waves * 3 = 24 N-tiles
    const int col = nt * 16 + l16;
    // B fragment base: B[k][n] = w[n][k]; lane holds col=l16, K chunk half*16
    const _Float16* bfp = fw16 + (size_t)col * DIMC + half * 16;
    const _Float16* bgp = gw16 + (size_t)col * DIMC + half * 16;

    v8f accf[4] = {};
    v8f accg[4] = {};
    // unroll_count(2): pipeline B global loads + A ds loads without the
    // full-unroll register explosion (round-1 asm spilled to scratch).
#pragma clang loop unroll_count(2)
    for (int kk = 0; kk < KSTEPS; ++kk) {
      const v16h bf = *(const v16h*)(bfp + kk * 32);
      const v16h bg = *(const v16h*)(bgp + kk * 32);
#pragma unroll
      for (int m = 0; m < 4; ++m) {
        // A fragment: lane row = l16, K chunk [half*16, +16)
        const _Float16* ap = &As[(m * 16 + l16) * SA + kk * 32 + half * 16];
        union { v16h v; v8h h[2]; } a;
        a.h[0] = *(const v8h*)ap;
        a.h[1] = *(const v8h*)(ap + 8);
        accf[m] = __builtin_amdgcn_wmma_f32_16x16x32_f16(
            false, a.v, false, bf, (short)0, accf[m], false, false);
        accg[m] = __builtin_amdgcn_wmma_f32_16x16x32_f16(
            false, a.v, false, bg, (short)0, accg[m], false, false);
      }
    }

    // ---- epilogue: bias, clip, exp, segment atomics (live in L2)
    const float fbias = fb[col];
    const float gbias = gb[col];
    float* Dcol = Dsum + col;
    float* Pcol = Psum + col;
#pragma unroll
    for (int m = 0; m < 4; ++m) {
#pragma unroll
      for (int r = 0; r < 8; ++r) {
        const int rloc = m * 16 + half * 8 + r;   // D layout: M = r + half*8
        const int seg  = segs[rloc];
        float g = accg[m][r] + gbias;
        g = fminf(fmaxf(g, -50.f), 50.f);
        const float w = __expf(g);
        const float f = accf[m][r] + fbias;
        atomicAdd(Dcol + (size_t)seg * DIMC, w);
        atomicAdd(Pcol + (size_t)seg * DIMC, f * w);
      }
    }
  }
}

// ------------------------------------------------------------ seg GEMM ------
// z = (Psum / max(Dsum,1e-9)) @ h_w^T + h_b   on [1024,384] — tiny.
__global__ void __launch_bounds__(256)
seg_gemm_kernel(const float* __restrict__ Dsum, const float* __restrict__ Psum,
                const _Float16* __restrict__ hw16, const float* __restrict__ hb,
                float* __restrict__ Z) {
  const int id   = blockIdx.x * 8 + (threadIdx.x >> 5);  // one wave = one tile
  const int lane = threadIdx.x & 31;
  const int st   = id / NTILES;
  const int nt   = id % NTILES;
  const int half = lane >> 4;
  const int l16  = lane & 15;
  const int col  = nt * 16 + l16;
  const int srow = st * 16 + l16;

  const _Float16* bp = hw16 + (size_t)col * DIMC + half * 16;
  const float*    Pp = Psum + (size_t)srow * DIMC + half * 16;
  const float*    Dp = Dsum + (size_t)srow * DIMC + half * 16;

  v8f acc = {};
#pragma clang loop unroll_count(2)
  for (int kk = 0; kk < KSTEPS; ++kk) {
    const v16h b = *(const v16h*)(bp + kk * 32);
    union { v16h v; _Float16 e[16]; } a;
#pragma unroll
    for (int j = 0; j < 8; ++j) {
      const float p0 = Pp[kk * 32 + j];
      const float d0 = Dp[kk * 32 + j];
      a.e[j] = (_Float16)(p0 / fmaxf(d0, 1e-9f));
      const float p1 = Pp[kk * 32 + 8 + j];
      const float d1 = Dp[kk * 32 + 8 + j];
      a.e[8 + j] = (_Float16)(p1 / fmaxf(d1, 1e-9f));
    }
    acc = __builtin_amdgcn_wmma_f32_16x16x32_f16(
        false, a.v, false, b, (short)0, acc, false, false);
  }

  const float bias = hb[col];
#pragma unroll
  for (int r = 0; r < 8; ++r) {
    const int s = st * 16 + half * 8 + r;
    Z[(size_t)s * DIMC + col] = acc[r] + bias;
  }
}

// -------------------------------------------------------------- gather ------
__global__ void __launch_bounds__(256)
gather_kernel(const int* __restrict__ ix, const float* __restrict__ Z,
              float* __restrict__ out) {
  const size_t i   = (size_t)blockIdx.x * 256 + threadIdx.x;  // H * (DIMC/4)
  const int    row = (int)(i / (DIMC / 4));
  const int    q   = (int)(i % (DIMC / 4));
  const int    seg = ix[row];
  const v4f v = ((const v4f*)Z)[(size_t)seg * (DIMC / 4) + q];
  __builtin_nontemporal_store(v, (v4f*)out + i);   // 96 MB write-once stream
}

// -------------------------------------------------------------- launch ------
extern "C" void kernel_launch(void* const* d_in, const int* in_sizes, int n_in,
                              void* d_out, int out_size, void* d_ws, size_t ws_size,
                              hipStream_t stream) {
  const float* x  = (const float*)d_in[0];
  const int*   ix = (const int*)d_in[1];
  const float* fw = (const float*)d_in[2];
  const float* fb = (const float*)d_in[3];
  const float* gw = (const float*)d_in[4];
  const float* gb = (const float*)d_in[5];
  const float* hw = (const float*)d_in[6];
  const float* hb = (const float*)d_in[7];
  float* out = (float*)d_out;

  const int H = in_sizes[0] / DIMC;  // 65536

  // workspace carve-up (~5.6 MB)
  float* Dsum = (float*)d_ws;
  float* Psum = Dsum + (size_t)NSEG * DIMC;
  float* Z    = Psum + (size_t)NSEG * DIMC;
  _Float16* fw16 = (_Float16*)(Z + (size_t)NSEG * DIMC);
  _Float16* gw16 = fw16 + (size_t)DIMC * DIMC;
  _Float16* hw16 = gw16 + (size_t)DIMC * DIMC;

  prep_kernel<<<512, 256, 0, stream>>>(fw, gw, hw, Dsum, Psum, fw16, gw16, hw16);

  main_kernel<<<H / MROWS, 256, 0, stream>>>(x, ix, fb, gb, fw16, gw16, Dsum, Psum);

  // 64 s-tiles * 24 n-tiles = 1536 wave-tiles, 8 waves/block -> 192 blocks
  seg_gemm_kernel<<<(NSEG / 16) * NTILES / 8, 256, 0, stream>>>(Dsum, Psum, hw16, hb, Z);

  gather_kernel<<<(int)((size_t)H * (DIMC / 4) / 256), 256, 0, stream>>>(ix, Z, out);
}
